// Transformer_19413252178504
// MI455X (gfx1250) — compile-verified, matrix operand order checked by reading, not run on hardware
//
#include <hip/hip_runtime.h>
#include <hip/hip_bf16.h>

// ------------------------------------------------------------------
// Transformer encoder-decoder forward on gfx1250 (CDNA5, wave32).
// All GEMMs run through v_wmma_f32_16x16x32_bf16 (f32 accumulate).
// Tile staging: b128 global loads -> v_perm-packed bf16 pairs -> b64 LDS stores.
// ------------------------------------------------------------------

#define DK      64
#define DMODEL  512
#define NHEAD   8
#define DFF2    2048
#define NLAYER  6
#define NB      2
#define SEQ     512
#define VOCAB   32000

typedef __attribute__((ext_vector_type(16))) __bf16 v16bf;
typedef __attribute__((ext_vector_type(8)))  __bf16 v8bf;
typedef __attribute__((ext_vector_type(8)))  float  v8f;

// Pack two floats into (bf16(a) | bf16(b) << 16) with round-to-nearest-even.
// Rounding add + single v_perm_b32 byte-merge: no 16-bit subregister ops.
__device__ __forceinline__ unsigned pack2bf(float a, float b) {
  union { float f; unsigned u; } x, y;
  x.f = a; y.f = b;
  unsigned ta = x.u + 0x7FFFu + ((x.u >> 16) & 1u);
  unsigned tb = y.u + 0x7FFFu + ((y.u >> 16) & 1u);
  // result bytes: [0]=ta.b2 [1]=ta.b3 [2]=tb.b2 [3]=tb.b3
  return __builtin_amdgcn_perm(tb, ta, 0x07060302u);
}

// ------------------------------------------------------------------
// Batched GEMM: C[z] = act(A[z] (MxK) * B[z] (KxN, or NxK if transB) + bias)
// 64x64 C tile per block of 128 threads (4 wave32 waves); BK = 32.
// LDS holds bf16 tiles in WMMA fragment layout: row-major, K-contiguous,
// B staged as B^T so A and B fragments load identically (2x ds_load_b128).
// ------------------------------------------------------------------
__global__ __launch_bounds__(128)
void gemm_bf16_wmma(const float* __restrict__ A, const float* __restrict__ Bm,
                    const float* __restrict__ bias, float* __restrict__ C,
                    int K, int lda, int ldb, int ldc,
                    long long oAo, long long oAi, long long oBo, long long oBi,
                    long long oCo, long long oCi, int zdiv,
                    int transB, int relu)
{
  __shared__ unsigned As32[64][16];   // [m][k/2] : 64 x 32 bf16
  __shared__ unsigned Bs32[64][16];   // [n][k/2] : 64 x 32 bf16 (B^T)
  const __bf16* Asb = (const __bf16*)&As32[0][0];
  const __bf16* Bsb = (const __bf16*)&Bs32[0][0];

  const int tid  = threadIdx.x;
  const int wave = tid >> 5;
  const int lane = tid & 31;
  const int mlan = lane & 15;
  const int kb   = (lane >> 4) << 3;   // 0 or 8: K sub-slot per lane half

  const int zo = blockIdx.z / zdiv;
  const int zi = blockIdx.z % zdiv;
  A  += (long long)zo * oAo + (long long)zi * oAi;
  Bm += (long long)zo * oBo + (long long)zi * oBi;
  C  += (long long)zo * oCo + (long long)zi * oCi;

  const int row0 = blockIdx.y * 64;
  const int col0 = blockIdx.x * 64;

  v8f acc[4];
  #pragma unroll
  for (int t = 0; t < 4; ++t) {
    float bv = bias ? bias[col0 + t * 16 + mlan] : 0.0f;
    #pragma unroll
    for (int r = 0; r < 8; ++r) acc[t][r] = bv;
  }

  for (int k0 = 0; k0 < K; k0 += 32) {
    __syncthreads();
    // --- stage A tile (64x32 f32): 4x global_load_b128 + packed b64 stores ---
    #pragma unroll
    for (int j = 0; j < 4; ++j) {
      int e4 = j * 128 + tid;          // 512 float4 covers 64x32
      int r  = e4 >> 3;                // 8 float4 per row
      int c4 = e4 & 7;
      const float4 f = *(const float4*)(A + (size_t)(row0 + r) * lda + k0 + c4 * 4);
      As32[r][c4 * 2 + 0] = pack2bf(f.x, f.y);
      As32[r][c4 * 2 + 1] = pack2bf(f.z, f.w);
    }
    // --- stage B tile as [n][k] ---
    if (transB) {                      // B is [N,K] row-major: same as A path
      #pragma unroll
      for (int j = 0; j < 4; ++j) {
        int e4 = j * 128 + tid;
        int r  = e4 >> 3;
        int c4 = e4 & 7;
        const float4 f = *(const float4*)(Bm + (size_t)(col0 + r) * ldb + k0 + c4 * 4);
        Bs32[r][c4 * 2 + 0] = pack2bf(f.x, f.y);
        Bs32[r][c4 * 2 + 1] = pack2bf(f.z, f.w);
      }
    } else {                           // B is [K,N]: 4x4 in-register transpose
      const int n4  = tid & 15;        // 16 column groups of 4
      const int kk4 = tid >> 4;        // 8 row groups of 4
      float fv[4][4];
      #pragma unroll
      for (int i = 0; i < 4; ++i)
        *(float4*)&fv[i][0] =
            *(const float4*)(Bm + (size_t)(k0 + kk4 * 4 + i) * ldb + col0 + n4 * 4);
      #pragma unroll
      for (int jn = 0; jn < 4; ++jn) {
        Bs32[n4 * 4 + jn][kk4 * 2 + 0] = pack2bf(fv[0][jn], fv[1][jn]);
        Bs32[n4 * 4 + jn][kk4 * 2 + 1] = pack2bf(fv[2][jn], fv[3][jn]);
      }
    }

    // --- prefetch next K tile while this one is consumed (global_prefetch_b8) ---
    if (k0 + 32 < K) {
      const int pr = tid >> 1;         // 0..63
      __builtin_prefetch(A + (size_t)(row0 + pr) * lda + (k0 + 32) + (tid & 1) * 16, 0, 3);
      const float* bq = transB
          ? (Bm + (size_t)(col0 + pr) * ldb + (k0 + 32) + (tid & 1) * 16)
          : (Bm + (size_t)(k0 + 32 + (tid >> 2)) * ldb + col0 + (tid & 3) * 16);
      __builtin_prefetch(bq, 0, 3);
    }
    __syncthreads();

    // A fragment: lane half selects K 0..7 / 8..15; elems 8..15 are K+16.
    const int arow = wave * 16 + mlan;
    v8bf alo = *(const v8bf*)(Asb + arow * 32 + kb);
    v8bf ahi = *(const v8bf*)(Asb + arow * 32 + 16 + kb);
    v16bf a;
    #pragma unroll
    for (int i = 0; i < 8; ++i) { a[i] = alo[i]; a[i + 8] = ahi[i]; }

    #pragma unroll
    for (int t = 0; t < 4; ++t) {
      const int brow = t * 16 + mlan;
      v8bf blo = *(const v8bf*)(Bsb + brow * 32 + kb);
      v8bf bhi = *(const v8bf*)(Bsb + brow * 32 + 16 + kb);
      v16bf b;
      #pragma unroll
      for (int i = 0; i < 8; ++i) { b[i] = blo[i]; b[i + 8] = bhi[i]; }
      acc[t] = __builtin_amdgcn_wmma_f32_16x16x32_bf16(
          false, a, false, b, (short)0, acc[t], false, false);
    }
  }

  // C/D layout: VGPR r -> M = (lane/16)*8 + r ; N = lane & 15
  #pragma unroll
  for (int t = 0; t < 4; ++t) {
    const int col = col0 + t * 16 + mlan;
    #pragma unroll
    for (int r = 0; r < 8; ++r) {
      const int row = row0 + wave * 16 + ((lane >> 4) << 3) + r;
      float v = acc[t][r];
      if (relu) v = fmaxf(v, 0.0f);
      C[(size_t)row * ldc + col] = v;
    }
  }
}

// ------------------------------------------------------------------
// Row softmax with optional causal mask. Mask applied BEFORE the 1/sqrt(DK)
// scale (matches reference: where(mask==0, -1e20, e) / 8).
// grid = (Lq, B*H), block = 256.
// ------------------------------------------------------------------
__global__ __launch_bounds__(256)
void softmax_kernel(float* __restrict__ P, int Lq, int Lk, int causal)
{
  const int row = blockIdx.x;
  const int tid = threadIdx.x;
  float* p = P + ((size_t)blockIdx.y * Lq + row) * Lk;
  __shared__ float red[256];
  const int per = Lk >> 8;             // Lk / 256 (Lk = 512 -> 2)
  float v[8];
  float lmax = -3.4e38f;
  for (int j = 0; j < per; ++j) {
    int k = (j << 8) + tid;
    float x = p[k];
    if (causal && k > row) x = -1.0e20f;
    x *= 0.125f;                       // 1/sqrt(64)
    v[j] = x;
    lmax = fmaxf(lmax, x);
  }
  red[tid] = lmax; __syncthreads();
  for (int s = 128; s > 0; s >>= 1) {
    if (tid < s) red[tid] = fmaxf(red[tid], red[tid + s]);
    __syncthreads();
  }
  const float m = red[0];
  __syncthreads();
  float lsum = 0.0f;
  for (int j = 0; j < per; ++j) { v[j] = __expf(v[j] - m); lsum += v[j]; }
  red[tid] = lsum; __syncthreads();
  for (int s = 128; s > 0; s >>= 1) {
    if (tid < s) red[tid] += red[tid + s];
    __syncthreads();
  }
  const float inv = 1.0f / red[0];
  for (int j = 0; j < per; ++j) p[(j << 8) + tid] = v[j] * inv;
}

// ------------------------------------------------------------------
// O[row] = LayerNorm(A[row] + B[row]) * g + b     (one block per row)
// ------------------------------------------------------------------
__global__ __launch_bounds__(256)
void add_ln_kernel(const float* __restrict__ A, const float* __restrict__ Bv,
                   const float* __restrict__ g, const float* __restrict__ b,
                   float* __restrict__ O, int Dm)
{
  const int row = blockIdx.x;
  const int tid = threadIdx.x;
  const size_t base = (size_t)row * Dm;
  __shared__ float r1[256], r2[256];
  const int per = Dm >> 8;             // Dm / 256 (512 -> 2)
  float x[8];
  float s = 0.0f, s2 = 0.0f;
  for (int j = 0; j < per; ++j) {
    int k = (j << 8) + tid;
    float v = A[base + k] + Bv[base + k];
    x[j] = v; s += v; s2 += v * v;
  }
  r1[tid] = s; r2[tid] = s2; __syncthreads();
  for (int st = 128; st > 0; st >>= 1) {
    if (tid < st) { r1[tid] += r1[tid + st]; r2[tid] += r2[tid + st]; }
    __syncthreads();
  }
  const float mean = r1[0] / (float)Dm;
  const float var  = r2[0] / (float)Dm - mean * mean;
  const float inv  = rsqrtf(var + 1e-5f);
  for (int j = 0; j < per; ++j) {
    int k = (j << 8) + tid;
    O[base + k] = (x[j] - mean) * inv * g[k] + b[k];
  }
}

// ------------------------------------------------------------------
// Token embedding gather + positional add.  grid = B*Slen.
// ------------------------------------------------------------------
__global__ __launch_bounds__(256)
void embed_kernel(const float* __restrict__ emb, const float* __restrict__ pos,
                  const int* __restrict__ tok, float* __restrict__ O,
                  int Slen, int Dm)
{
  const int bt = blockIdx.x;
  const int t  = bt % Slen;
  const int tk = tok[bt];
  const size_t ob = (size_t)bt * Dm;
  const size_t eb = (size_t)tk * Dm;
  const size_t pb = (size_t)t  * Dm;
  for (int d = threadIdx.x; d < Dm; d += blockDim.x)
    O[ob + d] = emb[eb + d] + pos[pb + d];
}

// ------------------------------------------------------------------
extern "C" void kernel_launch(void* const* d_in, const int* in_sizes, int n_in,
                              void* d_out, int out_size, void* d_ws, size_t ws_size,
                              hipStream_t stream)
{
  (void)in_sizes; (void)n_in; (void)out_size; (void)ws_size;

  const float* emb_src  = (const float*)d_in[0];
  const float* pos_src  = (const float*)d_in[1];
  const float* emb_tgt  = (const float*)d_in[2];
  const float* pos_tgt  = (const float*)d_in[3];
  const float* enc_wq   = (const float*)d_in[4];
  const float* enc_wk   = (const float*)d_in[5];
  const float* enc_wv   = (const float*)d_in[6];
  const float* enc_wo   = (const float*)d_in[7];
  const float* enc_wob  = (const float*)d_in[8];
  const float* dec_mwq  = (const float*)d_in[9];
  const float* dec_mwk  = (const float*)d_in[10];
  const float* dec_mwv  = (const float*)d_in[11];
  const float* dec_mwo  = (const float*)d_in[12];
  const float* dec_mwob = (const float*)d_in[13];
  const float* dec_wq   = (const float*)d_in[14];
  const float* dec_wk   = (const float*)d_in[15];
  const float* dec_wv   = (const float*)d_in[16];
  const float* dec_wo   = (const float*)d_in[17];
  const float* dec_wob  = (const float*)d_in[18];
  const float* enc_f1   = (const float*)d_in[19];
  const float* enc_f1b  = (const float*)d_in[20];
  const float* enc_f2   = (const float*)d_in[21];
  const float* enc_f2b  = (const float*)d_in[22];
  const float* enc_ln1s = (const float*)d_in[23];
  const float* enc_ln1b = (const float*)d_in[24];
  const float* enc_ln2s = (const float*)d_in[25];
  const float* enc_ln2b = (const float*)d_in[26];
  const float* dec_f1   = (const float*)d_in[27];
  const float* dec_f1b  = (const float*)d_in[28];
  const float* dec_f2   = (const float*)d_in[29];
  const float* dec_f2b  = (const float*)d_in[30];
  const float* dec_ln1s = (const float*)d_in[31];
  const float* dec_ln1b = (const float*)d_in[32];
  const float* dec_ln2s = (const float*)d_in[33];
  const float* dec_ln2b = (const float*)d_in[34];
  const float* dec_lns  = (const float*)d_in[35];
  const float* dec_lnb  = (const float*)d_in[36];
  const float* out_w    = (const float*)d_in[37];
  const float* out_b    = (const float*)d_in[38];
  const int*   src      = (const int*)d_in[39];
  const int*   tgt      = (const int*)d_in[40];
  float* out = (float*)d_out;

  // ---- workspace carve (all fp32), ~46 MB total ----
  float* ws = (float*)d_ws;
  size_t woff = 0;
  auto carve = [&](size_t n) { float* p = ws + woff; woff += n; return p; };
  const size_t ND = (size_t)NB * SEQ * DMODEL;
  float* xe  = carve(ND);                               // encoder activations
  float* y   = carve(ND);                               // decoder activations
  float* qb  = carve(ND);                               // decoder post-self-attn LN
  float* qh  = carve(ND);
  float* kh  = carve(ND);
  float* vh  = carve(ND);
  float* ctx = carve(ND);
  float* t0  = carve(ND);
  float* t1  = carve(ND);
  float* t2  = carve(ND);
  float* sc  = carve((size_t)NB * NHEAD * SEQ * SEQ);   // attention scores
  float* hff = carve((size_t)NB * SEQ * DFF2);          // FFN hidden

  auto gemm = [&](const float* Ap, const float* Bp, const float* biasp, float* Cp,
                  int M, int N, int Kp, int lda, int ldb, int ldc,
                  int nbatch, int zdiv,
                  long long oAo, long long oAi, long long oBo, long long oBi,
                  long long oCo, long long oCi, int transB, int relu) {
    dim3 g((unsigned)(N / 64), (unsigned)(M / 64), (unsigned)nbatch);
    gemm_bf16_wmma<<<g, 128, 0, stream>>>(Ap, Bp, biasp, Cp, Kp, lda, ldb, ldc,
                                          oAo, oAi, oBo, oBi, oCo, oCi, zdiv,
                                          transB, relu);
  };

  auto addln = [&](const float* a, const float* b, const float* g,
                   const float* bt, float* o) {
    add_ln_kernel<<<NB * SEQ, 256, 0, stream>>>(a, b, g, bt, o, DMODEL);
  };

  // Multi-head attention: per-head 64x64 projections -> QK^T -> softmax -> @V -> wo
  auto mha = [&](const float* qin, const float* kvin,
                 const float* wq, const float* wk, const float* wv,
                 const float* wo, const float* wob, int layer, int causal,
                 float* outp) {
    const float* wq_l  = wq  + (size_t)layer * DK * DK;
    const float* wk_l  = wk  + (size_t)layer * DK * DK;
    const float* wv_l  = wv  + (size_t)layer * DK * DK;
    const float* wo_l  = wo  + (size_t)layer * DMODEL * DMODEL;
    const float* wob_l = wob + (size_t)layer * DMODEL;
    const int rows = NB * SEQ * NHEAD;                  // flat [B*S*H, DK] view
    gemm(qin,  wq_l, nullptr, qh, rows, DK, DK, DK, DK, DK, 1, 1, 0,0,0,0,0,0, 0, 0);
    gemm(kvin, wk_l, nullptr, kh, rows, DK, DK, DK, DK, DK, 1, 1, 0,0,0,0,0,0, 0, 0);
    gemm(kvin, wv_l, nullptr, vh, rows, DK, DK, DK, DK, DK, 1, 1, 0,0,0,0,0,0, 0, 0);
    // scores[b,h] = qh[b,:,h,:] @ kh[b,:,h,:]^T   (both K-contiguous, transB=1)
    gemm(qh, kh, nullptr, sc, SEQ, SEQ, DK, DMODEL, DMODEL, SEQ,
         NB * NHEAD, NHEAD,
         (long long)SEQ * DMODEL, DK,
         (long long)SEQ * DMODEL, DK,
         (long long)NHEAD * SEQ * SEQ, (long long)SEQ * SEQ, 1, 0);
    softmax_kernel<<<dim3(SEQ, NB * NHEAD), 256, 0, stream>>>(sc, SEQ, SEQ, causal);
    // ctx[b,:,h,:] = probs[b,h] @ vh[b,:,h,:]
    gemm(sc, vh, nullptr, ctx, SEQ, DK, SEQ, SEQ, DMODEL, DMODEL,
         NB * NHEAD, NHEAD,
         (long long)NHEAD * SEQ * SEQ, (long long)SEQ * SEQ,
         (long long)SEQ * DMODEL, DK,
         (long long)SEQ * DMODEL, DK, 0, 0);
    // out = ctx @ wo + wob
    gemm(ctx, wo_l, wob_l, outp, NB * SEQ, DMODEL, DMODEL,
         DMODEL, DMODEL, DMODEL, 1, 1, 0,0,0,0,0,0, 0, 0);
  };

  auto ffn = [&](const float* f1, const float* f1b, const float* f2,
                 const float* f2b, const float* ln2s, const float* ln2b,
                 int layer, const float* xin, float* xout) {
    const float* f1_l  = f1  + (size_t)layer * DMODEL * DFF2;
    const float* f1b_l = f1b + (size_t)layer * DFF2;
    const float* f2_l  = f2  + (size_t)layer * DFF2 * DMODEL;
    const float* f2b_l = f2b + (size_t)layer * DMODEL;
    gemm(xin, f1_l, f1b_l, hff, NB * SEQ, DFF2, DMODEL,
         DMODEL, DFF2, DFF2, 1, 1, 0,0,0,0,0,0, 0, 1);          // ReLU
    gemm(hff, f2_l, f2b_l, t2, NB * SEQ, DMODEL, DFF2,
         DFF2, DMODEL, DMODEL, 1, 1, 0,0,0,0,0,0, 0, 0);
    addln(xin, t2, ln2s + (size_t)layer * DMODEL,
          ln2b + (size_t)layer * DMODEL, xout);
  };

  // ---------------- Encoder ----------------
  embed_kernel<<<NB * SEQ, 256, 0, stream>>>(emb_src, pos_src, src, xe, SEQ, DMODEL);
  for (int i = 0; i < NLAYER; ++i) {
    mha(xe, xe, enc_wq, enc_wk, enc_wv, enc_wo, enc_wob, i, 0, t0);
    addln(t0, xe, enc_ln1s + (size_t)i * DMODEL, enc_ln1b + (size_t)i * DMODEL, t1);
    ffn(enc_f1, enc_f1b, enc_f2, enc_f2b, enc_ln2s, enc_ln2b, i, t1, xe);
  }

  // ---------------- Decoder ----------------
  embed_kernel<<<NB * SEQ, 256, 0, stream>>>(emb_tgt, pos_tgt, tgt, y, SEQ, DMODEL);
  for (int i = 0; i < NLAYER; ++i) {
    mha(y, y, dec_mwq, dec_mwk, dec_mwv, dec_mwo, dec_mwob, i, 1, t0);   // causal
    addln(t0, y, dec_lns + (size_t)i * DMODEL, dec_lnb + (size_t)i * DMODEL, qb);
    mha(qb, xe, dec_wq, dec_wk, dec_wv, dec_wo, dec_wob, i, 0, t0);      // cross
    addln(t0, qb, dec_ln1s + (size_t)i * DMODEL, dec_ln1b + (size_t)i * DMODEL, t1);
    ffn(dec_f1, dec_f1b, dec_f2, dec_f2b, dec_ln2s, dec_ln2b, i, t1, y);
  }

  // ---------------- Logits: [B*T,512] x [512,32000] + out_b ----------------
  gemm(y, out_w, out_b, out, NB * SEQ, VOCAB, DMODEL,
       DMODEL, VOCAB, VOCAB, 1, 1, 0,0,0,0,0,0, 0, 0);
}
